// Mesh2MeshGNN_58171037057096
// MI455X (gfx1250) — matrix operand order repaired and checked by reading, not compiled, not run on hardware
//
#include <hip/hip_runtime.h>

// ---------------- types ----------------
typedef __bf16 bf16;
typedef bf16  v16bf __attribute__((ext_vector_type(16)));
typedef bf16  v8bf  __attribute__((ext_vector_type(8)));
typedef float v8f   __attribute__((ext_vector_type(8)));

#define D       128
#define NT      8        // N-tiles per 128 cols
#define KS1     8        // k-steps for K=256 (32 per step)
#define KS2     4        // k-steps for K=128
#define WAVES   8
#define BLOCK   256
#define TILE_E  128      // edges per block-tile (8 waves x 16 rows)

// fragment-element counts in LDS
#define WF1_ELEMS (NT*KS1*512)   // 32768 bf16 = 64 KB
#define WF2_ELEMS (NT*KS2*512)   // 16384 bf16 = 32 KB
#define HST_ELEMS (WAVES*16*D)   // 16384 bf16 = 32 KB
#define SMEM_BYTES ((WF1_ELEMS + WF2_ELEMS + HST_ELEMS) * 2)

__device__ __forceinline__ unsigned short f2bfu(float f) {
  unsigned int u = __builtin_bit_cast(unsigned int, f);
  u = (u + 0x7FFFu + ((u >> 16) & 1u)) >> 16;
  return (unsigned short)u;
}

__device__ __forceinline__ v16bf cat8(v8bf lo, v8bf hi) {
  return __builtin_shufflevector(lo, hi, 0,1,2,3,4,5,6,7,8,9,10,11,12,13,14,15);
}

// A-fragment: lane<16 reads K{0..7,16..23}, lane>=16 K{8..15,24..31} of its row.
// p must already include (row base + kbase + lhalf*8); chunks at +0 and +16 elems.
__device__ __forceinline__ v16bf loadA_bf(const unsigned short* p) {
  v8bf lo = *(const v8bf*)(p);
  v8bf hi = *(const v8bf*)(p + 16);
  return cat8(lo, hi);
}

// B-fragment packed in LDS: 16 contiguous bf16 per lane at frag*512 + lane*16.
__device__ __forceinline__ v16bf loadB_lds(const unsigned short* p) {
  v8bf lo = *(const v8bf*)(p);
  v8bf hi = *(const v8bf*)(p + 8);
  return cat8(lo, hi);
}

// A-fragment from fp32 source (agg rows in node kernel), converting to bf16.
__device__ __forceinline__ v16bf loadA_f32(const float* p) {
  v8f lo = *(const v8f*)(p);
  v8f hi = *(const v8f*)(p + 16);
  v16bf r;
#pragma unroll
  for (int i = 0; i < 8; i++) {
    r[i]     = __builtin_bit_cast(bf16, f2bfu(lo[i]));
    r[i + 8] = __builtin_bit_cast(bf16, f2bfu(hi[i]));
  }
  return r;
}

// Stage W [K x 128] (fp32, row-major) into LDS B-fragment layout (bf16).
// frag f = ntile*ksteps + kstep; element = f*512 + lane*16 + t
// (k = kstep*32 + (t>=8 ? 16:0) + (lane>=16 ? 8:0) + t%8, n = ntile*16 + lane%16)
__device__ __forceinline__ void stageW(unsigned short* dst, const float* W,
                                       int ksteps, int tid) {
  const int total = NT * ksteps * 512;
  for (int idx = tid; idx < total; idx += BLOCK) {
    int f = idx >> 9, r = idx & 511, l = r >> 4, t = r & 15;
    int ntile = f / ksteps, kstep = f % ksteps;
    int n = ntile * 16 + (l & 15);
    int k = kstep * 32 + ((t >> 3) << 4) + ((l >> 4) << 3) + (t & 7);
    dst[idx] = f2bfu(W[k * D + n]);
  }
}

// ---------------- prep: x -> bf16, agg -> 0 ----------------
__global__ void gnn_prep(const float* __restrict__ x,
                         unsigned short* __restrict__ xbf,
                         float* __restrict__ agg, int n) {
  int i = blockIdx.x * blockDim.x + threadIdx.x;
  if (i < n) { xbf[i] = f2bfu(x[i]); agg[i] = 0.0f; }
}

// ---------------- fused edge MLP + scatter-add ----------------
__global__ __launch_bounds__(BLOCK) void gnn_edge(
    const unsigned short* __restrict__ xbf,
    const long long* __restrict__ eidx,   // [2][E] int64
    const float* __restrict__ We1, const float* __restrict__ be1,
    const float* __restrict__ We2, const float* __restrict__ be2,
    float* __restrict__ agg, int E, int numTiles) {
  extern __shared__ char smem[];
  unsigned short* wf1 = (unsigned short*)smem;
  unsigned short* wf2 = wf1 + WF1_ELEMS;
  unsigned short* hst = wf2 + WF2_ELEMS;

  const int tid   = threadIdx.x;
  const int wave  = tid >> 5;
  const int lane  = tid & 31;
  const int l16   = lane & 15;
  const int lhalf = lane >> 4;

  stageW(wf1, We1, KS1, tid);
  stageW(wf2, We2, KS2, tid);
  __syncthreads();

  float b1[NT], b2[NT];
#pragma unroll
  for (int nt = 0; nt < NT; nt++) {
    b1[nt] = be1[nt * 16 + l16];
    b2[nt] = be2[nt * 16 + l16];
  }

  unsigned short* myh = hst + wave * (16 * D);   // private 16x128 bf16 tile

  for (int tile = blockIdx.x; tile < numTiles; tile += gridDim.x) {
    const int e0 = tile * TILE_E + wave * 16;
    int eL = e0 + l16; if (eL >= E) eL = E - 1;
    const int si = (int)eidx[eL];
    const int di = (int)eidx[(long long)E + eL];
    const unsigned short* srow = xbf + (size_t)si * D;
    const unsigned short* drow = xbf + (size_t)di * D;

    v8f acc[NT] = {};
#pragma unroll
    for (int ks = 0; ks < KS1; ks++) {
      const unsigned short* ap = (ks < 4) ? (srow + ks * 32 + lhalf * 8)
                                          : (drow + (ks - 4) * 32 + lhalf * 8);
      v16bf a = loadA_bf(ap);
#pragma unroll
      for (int nt = 0; nt < NT; nt++) {
        v16bf b = loadB_lds(wf1 + ((nt * KS1 + ks) << 9) + lane * 16);
        acc[nt] = __builtin_amdgcn_wmma_f32_16x16x32_bf16(
            false, a, false, b, (short)0, acc[nt], false, false);
      }
    }

    // bias + ReLU, C-layout -> bf16 LDS tile (in-order LDS within wave)
#pragma unroll
    for (int nt = 0; nt < NT; nt++) {
      const int n = nt * 16 + l16;
#pragma unroll
      for (int j = 0; j < 8; j++) {
        const int m = j + lhalf * 8;
        float v = acc[nt][j] + b1[nt];
        v = v > 0.0f ? v : 0.0f;
        myh[m * D + n] = f2bfu(v);
      }
    }

    v8f acc2[NT] = {};
#pragma unroll
    for (int ks = 0; ks < KS2; ks++) {
      v16bf a = loadA_bf(myh + l16 * D + ks * 32 + lhalf * 8);
#pragma unroll
      for (int nt = 0; nt < NT; nt++) {
        v16bf b = loadB_lds(wf2 + ((nt * KS2 + ks) << 9) + lane * 16);
        acc2[nt] = __builtin_amdgcn_wmma_f32_16x16x32_bf16(
            false, a, false, b, (short)0, acc2[nt], false, false);
      }
    }

    // scatter-add into L2-resident agg[dst] (fp32 atomics)
#pragma unroll
    for (int nt = 0; nt < NT; nt++) {
      const int n = nt * 16 + l16;
#pragma unroll
      for (int j = 0; j < 8; j++) {
        const int m = j + lhalf * 8;
        const float v = acc2[nt][j] + b2[nt];
        const int dm = __shfl(di, m, 32);
        if (e0 + m < E)
          unsafeAtomicAdd(&agg[(size_t)dm * D + n], v);
      }
    }
  }
}

// ---------------- node MLP + residual ----------------
__global__ __launch_bounds__(BLOCK) void gnn_node(
    const float* __restrict__ x, const unsigned short* __restrict__ xbf,
    const float* __restrict__ agg,
    const float* __restrict__ Wn1, const float* __restrict__ bn1,
    const float* __restrict__ Wn2, const float* __restrict__ bn2,
    float* __restrict__ out, int N) {
  extern __shared__ char smem[];
  unsigned short* wf1 = (unsigned short*)smem;
  unsigned short* wf2 = wf1 + WF1_ELEMS;
  unsigned short* hst = wf2 + WF2_ELEMS;

  const int tid   = threadIdx.x;
  const int wave  = tid >> 5;
  const int lane  = tid & 31;
  const int l16   = lane & 15;
  const int lhalf = lane >> 4;

  stageW(wf1, Wn1, KS1, tid);
  stageW(wf2, Wn2, KS2, tid);
  __syncthreads();

  float b1[NT], b2[NT];
#pragma unroll
  for (int nt = 0; nt < NT; nt++) {
    b1[nt] = bn1[nt * 16 + l16];
    b2[nt] = bn2[nt * 16 + l16];
  }

  unsigned short* myh = hst + wave * (16 * D);

  const int n0 = blockIdx.x * TILE_E + wave * 16;
  int rowL = n0 + l16; if (rowL >= N) rowL = N - 1;
  const unsigned short* xrow = xbf + (size_t)rowL * D;
  const float*          arow = agg + (size_t)rowL * D;

  v8f acc[NT] = {};
#pragma unroll
  for (int ks = 0; ks < KS1; ks++) {
    v16bf a = (ks < 4) ? loadA_bf(xrow + ks * 32 + lhalf * 8)
                       : loadA_f32(arow + (ks - 4) * 32 + lhalf * 8);
#pragma unroll
    for (int nt = 0; nt < NT; nt++) {
      v16bf b = loadB_lds(wf1 + ((nt * KS1 + ks) << 9) + lane * 16);
      acc[nt] = __builtin_amdgcn_wmma_f32_16x16x32_bf16(
          false, a, false, b, (short)0, acc[nt], false, false);
    }
  }

#pragma unroll
  for (int nt = 0; nt < NT; nt++) {
    const int n = nt * 16 + l16;
#pragma unroll
    for (int j = 0; j < 8; j++) {
      const int m = j + lhalf * 8;
      float v = acc[nt][j] + b1[nt];
      v = v > 0.0f ? v : 0.0f;
      myh[m * D + n] = f2bfu(v);
    }
  }

  v8f acc2[NT] = {};
#pragma unroll
  for (int ks = 0; ks < KS2; ks++) {
    v16bf a = loadA_bf(myh + l16 * D + ks * 32 + lhalf * 8);
#pragma unroll
    for (int nt = 0; nt < NT; nt++) {
      v16bf b = loadB_lds(wf2 + ((nt * KS2 + ks) << 9) + lane * 16);
      acc2[nt] = __builtin_amdgcn_wmma_f32_16x16x32_bf16(
          false, a, false, b, (short)0, acc2[nt], false, false);
    }
  }

#pragma unroll
  for (int nt = 0; nt < NT; nt++) {
    const int n = nt * 16 + l16;
#pragma unroll
    for (int j = 0; j < 8; j++) {
      const int m = j + lhalf * 8;
      const int node = n0 + m;
      if (node < N) {
        const size_t o = (size_t)node * D + n;
        out[o] = acc2[nt][j] + b2[nt] + x[o];
      }
    }
  }
}

// ---------------- launcher ----------------
extern "C" void kernel_launch(void* const* d_in, const int* in_sizes, int n_in,
                              void* d_out, int out_size, void* d_ws, size_t ws_size,
                              hipStream_t stream) {
  const float*     x    = (const float*)d_in[0];
  const long long* eidx = (const long long*)d_in[1];  // int64 [2, E]
  const float*     We1  = (const float*)d_in[2];
  const float*     be1  = (const float*)d_in[3];
  const float*     We2  = (const float*)d_in[4];
  const float*     be2  = (const float*)d_in[5];
  const float*     Wn1  = (const float*)d_in[6];
  const float*     bn1  = (const float*)d_in[7];
  const float*     Wn2  = (const float*)d_in[8];
  const float*     bn2  = (const float*)d_in[9];

  const int N = in_sizes[0] / D;       // 50000
  const int E = in_sizes[1] / 2;       // 800000
  const int nelem = N * D;

  // workspace: [xbf: N*D bf16][agg: N*D fp32]
  unsigned short* xbf = (unsigned short*)d_ws;
  size_t xbfBytes = ((size_t)nelem * 2 + 255) & ~(size_t)255;
  float* agg = (float*)((char*)d_ws + xbfBytes);

  gnn_prep<<<(nelem + BLOCK - 1) / BLOCK, BLOCK, 0, stream>>>(x, xbf, agg, nelem);

  const int numTiles = (E + TILE_E - 1) / TILE_E;       // 6250
  int nblk = numTiles < 1250 ? numTiles : 1250;         // 5 tiles/block
  gnn_edge<<<nblk, BLOCK, SMEM_BYTES, stream>>>(xbf, eidx, We1, be1, We2, be2,
                                                agg, E, numTiles);

  const int nodeTiles = (N + TILE_E - 1) / TILE_E;      // 391
  gnn_node<<<nodeTiles, BLOCK, SMEM_BYTES, stream>>>(x, xbf, agg, Wn1, bn1,
                                                     Wn2, bn2, (float*)d_out, N);
}
